// ChunkedLinearCrossEntropyLoss_19799799234621
// MI455X (gfx1250) — matrix operand order repaired and checked by reading, not compiled, not run on hardware
//
#include <hip/hip_runtime.h>
#include <stdint.h>

// Problem constants (match reference setup_inputs)
#define N_ROWS 4096
#define D_DIM  2048
#define V_DIM  32768
#define SOFTCAPF 20.0f
#define IGNORE_IDX (-100)
#define SMOOTH_A 0.1f
#define ZLW 1e-4f

// Tiling
#define COLGROUPS    16                        // grid.y : V partition
#define ROWS_PER_WG  256                       // 8 waves * 32 rows
#define WAVES_PER_WG 8
#define CPC          64                        // columns per chunk (4 col-tiles)
#define CHUNKS_PER_CG (V_DIM / CPC / COLGROUPS)  // 32
#define KSLAB        256                       // K staged per LDS slab
#define NSLABS       (D_DIM / KSLAB)           // 8
#define F_STRIDE     260                       // padded fp32 slab row (floats)
#define B_STRIDE     264                       // padded bf16 slab row (u16)

typedef __attribute__((ext_vector_type(16))) __bf16         v16bf;
typedef __attribute__((ext_vector_type(8)))  float          v8f;
typedef __attribute__((ext_vector_type(4)))  float          v4f;
typedef __attribute__((ext_vector_type(8)))  unsigned short v8u;
typedef __attribute__((ext_vector_type(4)))  unsigned int   v4ui;

static __device__ __forceinline__ unsigned pack_bf16_hi(unsigned u1,
                                                        unsigned u0) {
  // one v_perm_b32: [u1.b3,u1.b2,u0.b3,u0.b2] == packed {bf16(f1),bf16(f0)}
  return __builtin_amdgcn_perm(u1, u0, 0x07060302u);
}

// ---------------------------------------------------------------------------
// Pass 1: split fp32 hidden into bf16 hi/lo (raw u16) in workspace.
// hi = truncate-to-bf16(x) (exact subtraction), lo = trunc-bf16(x - hi).
// ---------------------------------------------------------------------------
__global__ void clce_split_hidden(const float* __restrict__ h,
                                  unsigned short* __restrict__ hi,
                                  unsigned short* __restrict__ lo, int n) {
  for (int i = blockIdx.x * blockDim.x + threadIdx.x; i < n;
       i += gridDim.x * blockDim.x) {
    const float f = h[i];
    const unsigned u = __float_as_uint(f);
    const float fh = __uint_as_float(u & 0xffff0000u);
    hi[i] = (unsigned short)(u >> 16);
    lo[i] = (unsigned short)(__float_as_uint(f - fh) >> 16);
  }
}

// ---------------------------------------------------------------------------
// Pass 2: fused GEMM (bf16x3 WMMA) + softcap + partial row statistics.
// grid = (N/256 row groups, COLGROUPS); 256 threads = 8 waves.
// Weight slab: async DMA (fp32 -> LDS), one cooperative hi/lo bf16 convert,
// pipelined so slab s+1 streams while slab s feeds the WMMAs.
// ---------------------------------------------------------------------------
__global__ __launch_bounds__(256, 1) void clce_gemm_stats(
    const unsigned short* __restrict__ hHi,
    const unsigned short* __restrict__ hLo,
    const float* __restrict__ W,
    const int* __restrict__ tgts,
    float* __restrict__ pS, float* __restrict__ pSl, float* __restrict__ pXt) {
  __shared__ float          Fs[CPC * F_STRIDE];  // fp32 slab [col][k]
  __shared__ unsigned short Hs[CPC * B_STRIDE];  // bf16 hi slab
  __shared__ unsigned short Ls[CPC * B_STRIDE];  // bf16 lo slab

  const int tid  = threadIdx.x;
  const int lane = tid & 31;
  const int wave = tid >> 5;
  const int lh   = lane >> 4;   // lane half (0/1)
  const int l16  = lane & 15;
  const int rowbase = blockIdx.x * ROWS_PER_WG + wave * 32;
  const int cg = blockIdx.y;

  // Targets for this wave's 32 rows (row = rowbase + rt*16 + lh*8 + v)
  int tg[2][8];
#pragma unroll
  for (int rt = 0; rt < 2; ++rt)
#pragma unroll
    for (int v = 0; v < 8; ++v)
      tg[rt][v] = tgts[rowbase + rt * 16 + lh * 8 + v];

  float sAcc[2][8], slAcc[2][8], xtAcc[2][8];
#pragma unroll
  for (int rt = 0; rt < 2; ++rt)
#pragma unroll
    for (int v = 0; v < 8; ++v) {
      sAcc[rt][v] = 0.f; slAcc[rt][v] = 0.f; xtAcc[rt][v] = 0.f;
    }

  // Low 32 bits of a generic LDS pointer == group-segment byte offset.
  const unsigned ldsF = (unsigned)(unsigned long long)(void*)Fs;
  const v8f v8zero = {0.f, 0.f, 0.f, 0.f, 0.f, 0.f, 0.f, 0.f};

  // Conversion-pass assignment: thread -> (col, 64-K segment).
  const int ccol = tid & 63;
  const int ckk0 = (tid >> 6) * 64;

  // Issue one 64-col x 256-K fp32 slab as async DMA into Fs.
  auto issue_slab = [&](int colbase, int kbase) {
#pragma unroll
    for (int j = 0; j < CPC / WAVES_PER_WG; ++j) {
      const int r = wave * (CPC / WAVES_PER_WG) + j;
      const unsigned ldsOff =
          ldsF + (unsigned)(r * F_STRIDE * 4) + (unsigned)(lane * 16);
      const unsigned long long ga =
          (unsigned long long)W +
          (((unsigned long long)(colbase + r) * D_DIM + kbase) * 4ull) +
          (unsigned long long)(lane * 16);
      asm volatile("global_load_async_to_lds_b128 %0, %1, off"
                   :: "v"(ldsOff), "v"(ga) : "memory");
      asm volatile("global_load_async_to_lds_b128 %0, %1, off"
                   :: "v"(ldsOff + 512u), "v"(ga + 512ull) : "memory");
    }
  };

  for (int ci = 0; ci < CHUNKS_PER_CG; ++ci) {
    const int colbase = (ci * COLGROUPS + cg) * CPC;
    v8f c[2][4];
#pragma unroll
    for (int rt = 0; rt < 2; ++rt)
#pragma unroll
      for (int ct = 0; ct < 4; ++ct) c[rt][ct] = v8zero;

    issue_slab(colbase, 0);

    for (int sl = 0; sl < NSLABS; ++sl) {
      asm volatile("s_wait_asynccnt 0" ::: "memory");
      __syncthreads();  // fp32 slab visible to all waves

      // Cooperative convert: Fs (fp32) -> Hs/Ls (packed bf16 hi/lo).
      {
        const float* fsrc = &Fs[ccol * F_STRIDE + ckk0];
        unsigned short* hdst = &Hs[ccol * B_STRIDE + ckk0];
        unsigned short* ldst = &Ls[ccol * B_STRIDE + ckk0];
#pragma unroll
        for (int i = 0; i < 8; ++i) {
          const v4f f0 = *(const v4f*)(fsrc + i * 8);
          const v4f f1 = *(const v4f*)(fsrc + i * 8 + 4);
          unsigned u[8];
          float fl[8];
#pragma unroll
          for (int e = 0; e < 4; ++e) {
            u[e]     = __float_as_uint(f0[e]);
            u[e + 4] = __float_as_uint(f1[e]);
          }
#pragma unroll
          for (int e = 0; e < 4; ++e) {
            fl[e]     = f0[e] - __uint_as_float(u[e] & 0xffff0000u);
            fl[e + 4] = f1[e] - __uint_as_float(u[e + 4] & 0xffff0000u);
          }
          v4ui hi, lo;
          hi.x = pack_bf16_hi(u[1], u[0]);
          hi.y = pack_bf16_hi(u[3], u[2]);
          hi.z = pack_bf16_hi(u[5], u[4]);
          hi.w = pack_bf16_hi(u[7], u[6]);
          lo.x = pack_bf16_hi(__float_as_uint(fl[1]), __float_as_uint(fl[0]));
          lo.y = pack_bf16_hi(__float_as_uint(fl[3]), __float_as_uint(fl[2]));
          lo.z = pack_bf16_hi(__float_as_uint(fl[5]), __float_as_uint(fl[4]));
          lo.w = pack_bf16_hi(__float_as_uint(fl[7]), __float_as_uint(fl[6]));
          *(v4ui*)(hdst + i * 8) = hi;
          *(v4ui*)(ldst + i * 8) = lo;
        }
      }
      __syncthreads();  // bf16 slabs ready; Fs free for next DMA

      if (sl + 1 < NSLABS) issue_slab(colbase, (sl + 1) * KSLAB);

      const int kbase = sl * KSLAB;
      for (int ks = 0; ks < KSLAB / 32; ++ks) {
        // --- A fragments: bf16 hi/lo straight from workspace (K-contiguous).
        // 16-bit A 16x32 layout: lanes 0-15 hold K=k0..k0+7 & k0+16..23,
        // lanes 16-31 hold K=k0+8..15 & k0+24..31 (ISA 7.12.2).
        v16bf ahi[2], alo[2];
#pragma unroll
        for (int rt = 0; rt < 2; ++rt) {
          const size_t rbo = (size_t)(rowbase + rt * 16 + l16) * D_DIM +
                             (size_t)(kbase + ks * 32 + lh * 8);
          const v8u h0 = *(const v8u*)(hHi + rbo);
          const v8u h1 = *(const v8u*)(hHi + rbo + 16);
          const v8u l0 = *(const v8u*)(hLo + rbo);
          const v8u l1 = *(const v8u*)(hLo + rbo + 16);
          ahi[rt] = __builtin_bit_cast(
              v16bf, __builtin_shufflevector(h0, h1, 0, 1, 2, 3, 4, 5, 6, 7, 8,
                                             9, 10, 11, 12, 13, 14, 15));
          alo[rt] = __builtin_bit_cast(
              v16bf, __builtin_shufflevector(l0, l1, 0, 1, 2, 3, 4, 5, 6, 7, 8,
                                             9, 10, 11, 12, 13, 14, 15));
        }
#pragma unroll
        for (int ct = 0; ct < 4; ++ct) {
          // --- B fragment: 16-bit B 32x16: lane n<16 -> col n, K=k0..k0+7 & k0+16..23;
          // lane n+16 -> col n, K=k0+8..15 & k0+24..31. Packed bf16 from LDS, no VALU.
          const int bo = (ct * 16 + l16) * B_STRIDE + ks * 32 + lh * 8;
          const v8u bh0 = *(const v8u*)(Hs + bo);
          const v8u bh1 = *(const v8u*)(Hs + bo + 16);
          const v8u bl0 = *(const v8u*)(Ls + bo);
          const v8u bl1 = *(const v8u*)(Ls + bo + 16);
          const v16bf bhi = __builtin_bit_cast(
              v16bf, __builtin_shufflevector(bh0, bh1, 0, 1, 2, 3, 4, 5, 6, 7,
                                             8, 9, 10, 11, 12, 13, 14, 15));
          const v16bf blo = __builtin_bit_cast(
              v16bf, __builtin_shufflevector(bl0, bl1, 0, 1, 2, 3, 4, 5, 6, 7,
                                             8, 9, 10, 11, 12, 13, 14, 15));
#pragma unroll
          for (int rt = 0; rt < 2; ++rt) {
            c[rt][ct] = __builtin_amdgcn_wmma_f32_16x16x32_bf16(
                false, ahi[rt], false, bhi, (short)0, c[rt][ct], false, false);
            c[rt][ct] = __builtin_amdgcn_wmma_f32_16x16x32_bf16(
                false, ahi[rt], false, blo, (short)0, c[rt][ct], false, false);
            c[rt][ct] = __builtin_amdgcn_wmma_f32_16x16x32_bf16(
                false, alo[rt], false, bhi, (short)0, c[rt][ct], false, false);
          }
        }
      }
    }

    // Softcap + statistics. |logit| <= 20 so exp(l-20) never overflows:
    // lse = 20 + log(sum exp(l-20)) -> no online max needed.
#pragma unroll
    for (int rt = 0; rt < 2; ++rt)
#pragma unroll
      for (int ct = 0; ct < 4; ++ct) {
        const int col = colbase + ct * 16 + l16;
#pragma unroll
        for (int v = 0; v < 8; ++v) {
          float l = c[rt][ct][v];
#if __has_builtin(__builtin_amdgcn_tanhf)
          l = SOFTCAPF * __builtin_amdgcn_tanhf(l * (1.0f / SOFTCAPF));
#else
          l = SOFTCAPF * tanhf(l * (1.0f / SOFTCAPF));
#endif
          sAcc[rt][v]  += __expf(l - SOFTCAPF);
          slAcc[rt][v] += l;
          if (col == tg[rt][v]) xtAcc[rt][v] += l;
        }
      }
  }

  // Row reductions over the 16 lanes holding each row, then per-colgroup
  // partial store (disjoint slices -> deterministic, no atomics).
#pragma unroll
  for (int rt = 0; rt < 2; ++rt)
#pragma unroll
    for (int v = 0; v < 8; ++v) {
      float s = sAcc[rt][v], sl = slAcc[rt][v], xt = xtAcc[rt][v];
#pragma unroll
      for (int m = 1; m < 16; m <<= 1) {
        s  += __shfl_xor(s,  m, 32);
        sl += __shfl_xor(sl, m, 32);
        xt += __shfl_xor(xt, m, 32);
      }
      if (l16 == 0) {
        const int row = rowbase + rt * 16 + lh * 8 + v;
        const size_t o = (size_t)cg * N_ROWS + row;
        pS[o] = s; pSl[o] = sl; pXt[o] = xt;
      }
    }
}

// ---------------------------------------------------------------------------
// Pass 3: combine colgroup partials -> per-row loss -> scalar (deterministic).
// ---------------------------------------------------------------------------
__global__ void clce_finalize(const float* __restrict__ pS,
                              const float* __restrict__ pSl,
                              const float* __restrict__ pXt,
                              const int* __restrict__ tgts,
                              float* __restrict__ out) {
  __shared__ float rL[256], rZ[256], rN[256];
  const int tid = threadIdx.x;
  float lossSum = 0.f, zSum = 0.f, nValid = 0.f;
  for (int row = tid; row < N_ROWS; row += 256) {
    float s = 0.f, sl = 0.f, xt = 0.f;
    for (int g = 0; g < COLGROUPS; ++g) {
      s  += pS [g * N_ROWS + row];
      sl += pSl[g * N_ROWS + row];
      xt += pXt[g * N_ROWS + row];
    }
    const int t = tgts[row];
    const float vf = (t != IGNORE_IDX) ? 1.f : 0.f;
    const float lse = SOFTCAPF + logf(s);
    const float nll = lse - xt;
    const float smooth = lse - sl * (1.0f / (float)V_DIM);
    const float rl = (1.0f - SMOOTH_A) * nll + SMOOTH_A * smooth;
    lossSum += rl * vf;
    const float z = lse * vf;
    zSum += z * z;
    nValid += vf;
  }
  rL[tid] = lossSum; rZ[tid] = zSum; rN[tid] = nValid;
  __syncthreads();
  for (int s = 128; s > 0; s >>= 1) {
    if (tid < s) {
      rL[tid] += rL[tid + s];
      rZ[tid] += rZ[tid + s];
      rN[tid] += rN[tid + s];
    }
    __syncthreads();
  }
  if (tid == 0) {
    const float nv = fmaxf(rN[0], 1.0f);
    out[0] = rL[0] / nv + ZLW * (rZ[0] / nv);
  }
}

// ---------------------------------------------------------------------------
extern "C" void kernel_launch(void* const* d_in, const int* in_sizes, int n_in,
                              void* d_out, int out_size, void* d_ws,
                              size_t ws_size, hipStream_t stream) {
  const float* hidden  = (const float*)d_in[0];  // [N, D] fp32
  const float* weight  = (const float*)d_in[1];  // [V, D] fp32
  const int*   targets = (const int*)d_in[2];    // [N] int32

  // Workspace layout (~34.3 MB):
  //   hHi  : N*D u16 (bf16 hi)     16 MB
  //   hLo  : N*D u16 (bf16 lo)     16 MB
  //   pS/pSl/pXt : COLGROUPS*N f32 each (768 KB total)
  unsigned short* hHi = (unsigned short*)d_ws;
  unsigned short* hLo = hHi + (size_t)N_ROWS * D_DIM;
  float* pS  = (float*)(hLo + (size_t)N_ROWS * D_DIM);
  float* pSl = pS + (size_t)COLGROUPS * N_ROWS;
  float* pXt = pSl + (size_t)COLGROUPS * N_ROWS;

  clce_split_hidden<<<2048, 256, 0, stream>>>(hidden, hHi, hLo,
                                              N_ROWS * D_DIM);
  dim3 grid(N_ROWS / ROWS_PER_WG, COLGROUPS);
  clce_gemm_stats<<<grid, 256, 0, stream>>>(hHi, hLo, weight, targets, pS, pSl,
                                            pXt);
  clce_finalize<<<1, 256, 0, stream>>>(pS, pSl, pXt, targets, (float*)d_out);
}
